// FusionAfterBEVSEDirect_23072564314129
// MI455X (gfx1250) — matrix operand order repaired and checked by reading, not compiled, not run on hardware
//
#include <hip/hip_runtime.h>
#include <hip/hip_bf16.h>
#include <math.h>
#include <stdint.h>

// ---------------- problem constants ----------------
#define B_   2
#define H_   256
#define Wd_  256
#define C_   128
#define CI_  128
#define CR_  64
#define NH_  8
#define DH_  16
#define DI_  256
#define DS_  16
#define KTOP_ 128
#define NROW_ (B_*H_)            // 512
#define NW_   ((long)NROW_*Wd_)  // 131072 pixel rows

typedef _Float16 v16h __attribute__((ext_vector_type(16)));
typedef float    v8f  __attribute__((ext_vector_type(8)));
typedef unsigned int v4u __attribute__((ext_vector_type(4)));
typedef int      v4i  __attribute__((ext_vector_type(4)));
typedef int      v8i  __attribute__((ext_vector_type(8)));

#ifndef __has_builtin
#define __has_builtin(x) 0
#endif
#if __has_builtin(__builtin_amdgcn_tensor_load_to_lds)
#define HAVE_TDM 1
#else
#define HAVE_TDM 0
#endif
#if __has_builtin(__builtin_amdgcn_s_wait_tensorcnt)
#define TDM_WAIT() __builtin_amdgcn_s_wait_tensorcnt(0)
#else
#define TDM_WAIT() ((void)0)
#endif

// ---------------- WMMA fragment helpers (CDNA5, wave32) ----------------
// A-matrix 16x32 f16 layout (ISA 7.12.2): lanes 0-15 -> K 0..7 & 16..23,
// lanes 16-31 -> K 8..15 & 24..31; element i -> k = kb + (i&7) + 16*(i>>3).
__device__ __forceinline__ v16h frag_a(const _Float16* p, int ms, int ks, int kmax) {
  int lane = threadIdx.x & 31;
  int m  = lane & 15;
  int kb = (lane >> 4) << 3;
  v16h v;
#pragma unroll
  for (int i = 0; i < 16; ++i) {
    int k = kb + (i & 7) + ((i >> 3) << 4);
    v[i] = (k < kmax) ? p[m * ms + k * ks] : (_Float16)0.0f;
  }
  return v;
}
// B-matrix 32x16 f16: same K order as A: lanes 0-15 -> K 0..7 & 16..23, lanes 16-31 -> K 8..15 & 24..31.
__device__ __forceinline__ v16h frag_b(const _Float16* p, int ns, int ks, int kmax) {
  int lane = threadIdx.x & 31;
  int n  = lane & 15;
  int kb = (lane >> 4) << 3;
  v16h v;
#pragma unroll
  for (int i = 0; i < 16; ++i) {
    int k = kb + (i & 7) + ((i >> 3) << 4);
    v[i] = (k < kmax) ? p[n * ns + k * ks] : (_Float16)0.0f;
  }
  return v;
}
__device__ __forceinline__ v8f wmma32(v16h a, v16h b, v8f c) {
  return __builtin_amdgcn_wmma_f32_16x16x32_f16(false, a, false, b, (short)0, c,
                                                false, false);
}

#if HAVE_TDM
// TDM 2D tile load: 64 rows x 64 f16 tile, row stride = strideElems (f16 units),
// into LDS at ldsOff with 66-element pitch via D# padding
// (pad_interval code 4 = every 32 DWORDs = one 128B row, pad_amount code 0 = 1 DWORD = 2 f16).
__device__ __forceinline__ void tdm_load_2d(const void* gptr, unsigned ldsOff, int strideElems) {
  unsigned long long ga = (unsigned long long)(uintptr_t)gptr;
  v4u g0;
  g0[0] = 1u;                                        // count=1, user descriptor
  g0[1] = ldsOff;                                    // lds_addr (bytes)
  g0[2] = (unsigned)(ga & 0xFFFFFFFFull);            // global_addr[31:0]
  g0[3] = (unsigned)((ga >> 32) & 0x01FFFFFFull) | (2u << 30);  // global_addr[56:32] | type=2
  v8i g1;
  g1[0] = (int)((1u << 16) | (1u << 20) | (4u << 22));          // data_size=2B, pad_enable, interval=32DW (+pad_amount=1DW)
  g1[1] = (int)((64u & 0xFFFFu) << 16);              // tensor_dim0 lo (=64)
  g1[2] = (int)(((64u >> 16) & 0xFFFFu) | ((64u & 0xFFFFu) << 16)); // tensor_dim0 hi | tensor_dim1 lo (=64)
  g1[3] = (int)(((64u >> 16) & 0xFFFFu) | ((64u & 0xFFFFu) << 16)); // tensor_dim1 hi | tile_dim0 (=64)
  g1[4] = (int)(64u & 0xFFFFu);                      // tile_dim1 (=64), tile_dim2=0
  g1[5] = (int)(unsigned)strideElems;                // tensor_dim0_stride lo32
  g1[6] = 0;
  g1[7] = 0;
  v4i z = {0, 0, 0, 0};
#if __clang_major__ >= 23
  v8i z8 = {0, 0, 0, 0, 0, 0, 0, 0};
  __builtin_amdgcn_tensor_load_to_lds(g0, g1, z, z, z8, 0);
#else
  __builtin_amdgcn_tensor_load_to_lds(g0, g1, z, z, 0);
#endif
}
#endif

// ---------------- small utility kernels ----------------
__global__ void cvt16_kernel(const float* __restrict__ src, _Float16* __restrict__ dst, long n) {
  long i = (long)blockIdx.x * blockDim.x + threadIdx.x;
  if (i < n) dst[i] = (_Float16)src[i];
}

// (B,Cn,H,W) f32 -> (N*W, Cn) f16 sequence layout, row = (b*H+h)*W + w
__global__ void chw_to_seq16(const float* __restrict__ in, _Float16* __restrict__ out, int Cn) {
  long i = (long)blockIdx.x * blockDim.x + threadIdx.x;
  long total = (long)B_ * Cn * H_ * Wd_;
  if (i >= total) return;
  int w = (int)(i & 255);
  long t = i >> 8;
  int h = (int)(t & 255);
  t >>= 8;
  int c = (int)(t % Cn);
  int b = (int)(t / Cn);
  out[(((long)(b * H_ + h) * Wd_) + w) * Cn + c] = (_Float16)in[i];
}

// per-row LayerNorm over 128 channels, f32 in -> f16 out
__global__ void ln128_kernel(const float* __restrict__ x, const float* __restrict__ g,
                             const float* __restrict__ bta, _Float16* __restrict__ out, long M) {
  long r = (long)blockIdx.x * blockDim.x + threadIdx.x;
  if (r >= M) return;
  const float* p = x + r * 128;
  float m = 0.f;
  for (int i = 0; i < 128; ++i) m += p[i];
  m *= (1.f / 128.f);
  float v = 0.f;
  for (int i = 0; i < 128; ++i) { float d = p[i] - m; v += d * d; }
  v *= (1.f / 128.f);
  float inv = rsqrtf(v + 1e-5f);
  _Float16* o = out + r * 128;
  for (int i = 0; i < 128; ++i) o[i] = (_Float16)((p[i] - m) * inv * g[i] + bta[i]);
}

// ---------------- generic 64x64-tile WMMA GEMM ----------------
// out[M,Nout] = act( [A | A2][M, K1+K2] * Wt[Nout, K]^T + bias )
// flags: 1=relu  2=softplus  4=accumulate into outF
// K staged 64 at a time; A tile double-buffered via the Tensor Data Mover when
// the row is contiguous (A2==null); B tile loaded by the waves in parallel.
__global__ void gemm64x64(const _Float16* __restrict__ A, int K1,
                          const _Float16* __restrict__ A2, int K2,
                          const _Float16* __restrict__ Wt,
                          const float* __restrict__ bias,
                          float* __restrict__ outF, _Float16* __restrict__ outH,
                          long M, int Nout, int flags) {
  __shared__ _Float16 As[2][64][66];
  __shared__ _Float16 Bs[64][66];
  const int K = K1 + K2;
  long m0 = (long)blockIdx.x * 64;
  int n0 = blockIdx.y * 64;
  int tid = threadIdx.x;
  int wave = tid >> 5;
  int mt = wave & 3;
  int ntA = wave >> 2;
  int ntB = ntA + 2;
  v8f acc0 = {};
  v8f acc1 = {};
#if HAVE_TDM
  const bool useTdm = (A2 == nullptr) && (m0 + 64 <= M) && ((K & 63) == 0);
  if (useTdm && wave == 0)
    tdm_load_2d(A + m0 * K, (unsigned)(uintptr_t)&As[0][0][0], K);
#else
  const bool useTdm = false;
#endif
  int cur = 0;
  for (int kc = 0; kc < K; kc += 64) {
    // stage B (all waves) while the TDM streams A
#pragma unroll
    for (int j = 0; j < 16; ++j) {
      int e = tid + j * 256;             // 4096 = 64x64
      int mm = e >> 6, kk = e & 63;
      int ng = n0 + mm;
      Bs[mm][kk] = (ng < Nout) ? Wt[(long)ng * K + kc + kk] : (_Float16)0.0f;
    }
    if (useTdm) {
      if (wave == 0) TDM_WAIT();
    } else {
#pragma unroll
      for (int j = 0; j < 16; ++j) {
        int e = tid + j * 256;
        int mm = e >> 6, kk = e & 63;
        int col = kc + kk;
        long mg = m0 + mm;
        _Float16 av = (_Float16)0.0f;
        if (mg < M)
          av = (col < K1) ? A[mg * K1 + col] : A2[mg * K2 + (col - K1)];
        As[cur][mm][kk] = av;
      }
    }
    __syncthreads();
#if HAVE_TDM
    if (useTdm && wave == 0 && kc + 64 < K)   // prefetch next chunk into other buffer
      tdm_load_2d(A + m0 * K + kc + 64, (unsigned)(uintptr_t)&As[cur ^ 1][0][0], K);
#endif
#pragma unroll
    for (int ks = 0; ks < 64; ks += 32) {
      v16h a  = frag_a(&As[cur][mt * 16][ks], 66, 1, 32);
      v16h b0 = frag_b(&Bs[ntA * 16][ks], 66, 1, 32);
      v16h b1 = frag_b(&Bs[ntB * 16][ks], 66, 1, 32);
      acc0 = wmma32(a, b0, acc0);
      acc1 = wmma32(a, b1, acc1);
    }
    __syncthreads();
    cur ^= 1;
  }
  int lane = tid & 31;
  int nn = lane & 15;
  int mb = (lane >> 4) << 3;
#pragma unroll
  for (int v = 0; v < 8; ++v) {
    long mg = m0 + mt * 16 + mb + v;
    if (mg >= M) continue;
#pragma unroll
    for (int s = 0; s < 2; ++s) {
      int ng = n0 + (s == 0 ? ntA : ntB) * 16 + nn;
      if (ng >= Nout) continue;
      float x = (s == 0 ? acc0[v] : acc1[v]) + (bias ? bias[ng] : 0.f);
      if (flags & 1) x = fmaxf(x, 0.f);
      if (flags & 2) x = (x > 20.f) ? x : logf(1.f + __expf(x));
      long o = mg * Nout + ng;
      if (outF) { if (flags & 4) outF[o] += x; else outF[o] = x; }
      if (outH) outH[o] = (_Float16)x;
    }
  }
}

// ---------------- score -> exact top-k (rank counting) ----------------
__global__ void score_topk(const _Float16* __restrict__ h1, const float* __restrict__ Ws2,
                           const float* __restrict__ bs2, int* __restrict__ idx) {
  __shared__ float sc[256];
  __shared__ float w2[64];
  int n = blockIdx.x;
  int t = threadIdx.x;
  if (t < 64) w2[t] = Ws2[t];
  __syncthreads();
  const _Float16* p = h1 + ((long)n * Wd_ + t) * 64;
  float s = bs2[0];
  for (int i = 0; i < 64; ++i) s += (float)p[i] * w2[i];
  sc[t] = s;
  __syncthreads();
  int rank = 0;
  for (int j = 0; j < 256; ++j) {
    float sj = sc[j];
    if (sj > s || (sj == s && j < t)) ++rank;
  }
  if (rank < KTOP_) idx[n * KTOP_ + rank] = t;
}

// gather top-k image rows + LayerNorm -> kvln (N,KTOP,128) f16
__global__ void gather_ln_kv(const _Float16* __restrict__ imgseq, const int* __restrict__ idx,
                             const float* __restrict__ g, const float* __restrict__ bta,
                             _Float16* __restrict__ kvout) {
  int n = blockIdx.x;
  int j = threadIdx.x;
  if (j >= KTOP_) return;
  int row = idx[n * KTOP_ + j];
  const _Float16* p = imgseq + ((long)n * Wd_ + row) * 128;
  float m = 0.f;
  for (int i = 0; i < 128; ++i) m += (float)p[i];
  m *= (1.f / 128.f);
  float v = 0.f;
  for (int i = 0; i < 128; ++i) { float d = (float)p[i] - m; v += d * d; }
  v *= (1.f / 128.f);
  float inv = rsqrtf(v + 1e-5f);
  _Float16* o = kvout + ((long)n * KTOP_ + j) * 128;
  for (int i = 0; i < 128; ++i) o[i] = (_Float16)(((float)p[i] - m) * inv * g[i] + bta[i]);
}

// ---------------- attention core: softmax(QK^T/4)V per head ----------------
__global__ void attn_core(const _Float16* __restrict__ qh, const _Float16* __restrict__ kh,
                          const _Float16* __restrict__ vh, _Float16* __restrict__ ao) {
  extern __shared__ char smem[];
  _Float16* qs  = (_Float16*)smem;                 // [64][132]
  _Float16* Sh  = qs + 64 * 132;                   // [64][132]
  float*    Sf  = (float*)(Sh + 64 * 132);         // [64][132]
  float*    aof = Sf + 64 * 132;                   // [64][128]
  _Float16* ks  = (_Float16*)(aof + 64 * 128);     // [128][18]
  _Float16* vs  = ks + 128 * 18;                   // [128][18]
  int n = blockIdx.x;
  int q0 = blockIdx.y * 64;
  int tid = threadIdx.x;
  int wave = tid >> 5;
  for (int e = tid; e < 64 * 128; e += 256) {
    int m = e >> 7, c = e & 127;
    qs[m * 132 + c] = qh[((long)n * Wd_ + q0 + m) * 128 + c];
    aof[e] = 0.f;
  }
  __syncthreads();
  for (int h = 0; h < NH_; ++h) {
    for (int e = tid; e < KTOP_ * 16; e += 256) {
      int kr = e >> 4, d = e & 15;
      long base = ((long)n * KTOP_ + kr) * 128 + h * 16 + d;
      ks[kr * 18 + d] = kh[base];
      vs[kr * 18 + d] = vh[base];
    }
    __syncthreads();
    // S = Q_h(64x16) * K_h^T(16x128), K padded 16->32 with zeros
    {
      int mt = wave & 3;
      int ntb = (wave >> 2) * 4;
      int lane = tid & 31;
      int nn = lane & 15;
      int mb = (lane >> 4) << 3;
#pragma unroll
      for (int j = 0; j < 4; ++j) {
        int nt = ntb + j;
        v16h a = frag_a(&qs[(mt * 16) * 132 + h * 16], 132, 1, 16);
        v16h b = frag_b(&ks[(nt * 16) * 18], 18, 1, 16);
        v8f acc = {};
        acc = wmma32(a, b, acc);
#pragma unroll
        for (int v = 0; v < 8; ++v)
          Sf[(mt * 16 + mb + v) * 132 + nt * 16 + nn] = acc[v] * 0.25f; // 1/sqrt(DH)
      }
    }
    __syncthreads();
    if (tid < 64) {
      float mx = -1e30f;
      for (int k = 0; k < KTOP_; ++k) mx = fmaxf(mx, Sf[tid * 132 + k]);
      float s = 0.f;
      for (int k = 0; k < KTOP_; ++k) { float e = __expf(Sf[tid * 132 + k] - mx); Sf[tid * 132 + k] = e; s += e; }
      float inv = 1.f / s;
      for (int k = 0; k < KTOP_; ++k) Sh[tid * 132 + k] = (_Float16)(Sf[tid * 132 + k] * inv);
    }
    __syncthreads();
    // O_h = S(64x128) * V_h(128x16)
    if (wave < 4) {
      int mt = wave;
      v8f acc = {};
      for (int kc = 0; kc < KTOP_; kc += 32) {
        v16h a = frag_a(&Sh[(mt * 16) * 132 + kc], 132, 1, 32);
        v16h b = frag_b(&vs[kc * 18], 1, 18, 32);
        acc = wmma32(a, b, acc);
      }
      int lane = tid & 31;
      int nn = lane & 15;
      int mb = (lane >> 4) << 3;
#pragma unroll
      for (int v = 0; v < 8; ++v)
        aof[(mt * 16 + mb + v) * 128 + h * 16 + nn] = acc[v];
    }
    __syncthreads();
  }
  for (int e = tid; e < 64 * 128; e += 256) {
    int m = e >> 7, c = e & 127;
    ao[((long)n * Wd_ + q0 + m) * 128 + c] = (_Float16)aof[m * 128 + c];
  }
}

// ---------------- causal depthwise conv (DC=4) + SiLU ----------------
__global__ void dwconv_silu(const _Float16* __restrict__ xz, const float* __restrict__ cdw,
                            const float* __restrict__ bdw, _Float16* __restrict__ xs) {
  long i = (long)blockIdx.x * 256 + threadIdx.x;
  if (i >= NW_ * (long)DI_) return;
  int d = (int)(i & 255);
  long nw = i >> 8;
  int w = (int)(nw & 255);
  long n = nw >> 8;
  float acc = bdw[d];
#pragma unroll
  for (int j = 0; j < 4; ++j) {
    int ww = w - 3 + j;
    if (ww >= 0) acc += (float)xz[((long)n * Wd_ + ww) * 512 + d] * cdw[d * 4 + j];
  }
  xs[i] = (_Float16)(acc / (1.f + __expf(-acc)));
}

// ---------------- bidirectional selective scan + gate by SiLU(z) ----------------
__global__ void ssm_scan(const _Float16* __restrict__ dt, const _Float16* __restrict__ xs,
                         const float* __restrict__ Btg, const float* __restrict__ Ctg,
                         const float* __restrict__ A_log, const float* __restrict__ Dp,
                         const _Float16* __restrict__ xz, _Float16* __restrict__ ybuf) {
  __shared__ float Bs[Wd_ * DS_];
  __shared__ float Cs[Wd_ * DS_];
  int n = blockIdx.x;
  int d = threadIdx.x;
  for (int e = d; e < Wd_ * DS_; e += 256) {
    Bs[e] = Btg[(long)n * Wd_ * DS_ + e];
    Cs[e] = Ctg[(long)n * Wd_ * DS_ + e];
  }
  __syncthreads();
  float Ar[DS_];
#pragma unroll
  for (int s = 0; s < DS_; ++s) Ar[s] = -__expf(A_log[d * DS_ + s]);
  float Dd = Dp[d];
  float h[DS_];
#pragma unroll
  for (int s = 0; s < DS_; ++s) h[s] = 0.f;
  for (int w = 0; w < Wd_; ++w) {            // forward
    long base = (long)n * Wd_ + w;
    float dtv = (float)dt[base * DI_ + d];
    float xv  = (float)xs[base * DI_ + d];
    float y = 0.f;
#pragma unroll
    for (int s = 0; s < DS_; ++s) {
      h[s] = __expf(dtv * Ar[s]) * h[s] + dtv * xv * Bs[w * DS_ + s];
      y += h[s] * Cs[w * DS_ + s];
    }
    ybuf[base * DI_ + d] = (_Float16)y;
  }
#pragma unroll
  for (int s = 0; s < DS_; ++s) h[s] = 0.f;
  for (int w = Wd_ - 1; w >= 0; --w) {       // backward + finalize
    long base = (long)n * Wd_ + w;
    float dtv = (float)dt[base * DI_ + d];
    float xv  = (float)xs[base * DI_ + d];
    float y = 0.f;
#pragma unroll
    for (int s = 0; s < DS_; ++s) {
      h[s] = __expf(dtv * Ar[s]) * h[s] + dtv * xv * Bs[w * DS_ + s];
      y += h[s] * Cs[w * DS_ + s];
    }
    float tot = (float)ybuf[base * DI_ + d] + y + Dd * xv;
    float z = (float)xz[base * 512 + DI_ + d];
    tot *= z / (1.f + __expf(-z));
    ybuf[base * DI_ + d] = (_Float16)tot;
  }
}

// ---------------- sigmoid gate + height mask + enhanced radar ----------------
__global__ void gate_enh(const float* __restrict__ radar, const float* __restrict__ img,
                         const float* __restrict__ Wg, const float* __restrict__ bg,
                         const float* __restrict__ gamma, const float* __restrict__ gammah,
                         const float* __restrict__ dA, const float* __restrict__ dH,
                         float* __restrict__ renh) {
  extern __shared__ char smem[];
  float* wgs = (float*)smem;          // [64][192]
  float* pix = wgs + 64 * 192;        // [64][192]
  int bh = blockIdx.x;
  int b = bh >> 8, h = bh & 255;
  int w0 = blockIdx.y * 64;
  int tid = threadIdx.x;
  for (int e = tid; e < 64 * 192; e += 256) wgs[e] = Wg[e];
  for (int e = tid; e < 64 * 192; e += 256) {
    int c = e >> 6, p = e & 63;
    pix[p * 192 + c] = (c < 64)
        ? radar[(((long)b * 64 + c) * H_ + h) * Wd_ + w0 + p]
        : img[(((long)b * 128 + (c - 64)) * H_ + h) * Wd_ + w0 + p];
  }
  __syncthreads();
  float wm = fmaxf(1.0f - 0.7f * (float)h / (float)(H_ - 1), 0.3f);
  float g = gamma[0], gh = gammah[0];
  int p = tid & 63;
  int qg = tid >> 6;
  long nw = (long)bh * Wd_ + w0 + p;
  for (int cr = qg * 16; cr < qg * 16 + 16; ++cr) {
    float acc = bg[cr];
    for (int c = 0; c < 192; ++c) acc += wgs[cr * 192 + c] * pix[p * 192 + c];
    float gate = wm / (1.f + __expf(-acc));
    long o = (((long)b * 64 + cr) * H_ + h) * Wd_ + w0 + p;
    renh[o] = radar[o] + g * gate * dA[nw * 64 + cr] + gh * gate * dH[nw * 64 + cr];
  }
}

// ---------------- 3x3 fuse conv as implicit WMMA GEMM + BN + ReLU ----------------
__global__ void conv3x3_bn_relu(const float* __restrict__ renh, const float* __restrict__ img,
                                const _Float16* __restrict__ wf, const float* __restrict__ bng,
                                const float* __restrict__ bnb, const float* __restrict__ bnm,
                                const float* __restrict__ bnv, float* __restrict__ out) {
  extern __shared__ char smem[];
  _Float16* ain = (_Float16*)smem;       // [3][192][68] (66 used)
  _Float16* Bsw = ain + 3 * 192 * 68;    // [64][34]
  int bh = blockIdx.x;
  int b = bh >> 8, h = bh & 255;
  int w0 = blockIdx.y * 64;
  int n0 = blockIdx.z * 64;
  int tid = threadIdx.x;
  for (int e = tid; e < 3 * 192 * 66; e += 256) {
    int wl = e % 66;
    int rest = e / 66;
    int ci = rest % 192;
    int dh = rest / 192;
    int hh = h + dh - 1;
    int ww = w0 - 1 + wl;
    float v = 0.f;
    if (hh >= 0 && hh < H_ && ww >= 0 && ww < Wd_)
      v = (ci < 64) ? renh[(((long)b * 64 + ci) * H_ + hh) * Wd_ + ww]
                    : img[(((long)b * 128 + (ci - 64)) * H_ + hh) * Wd_ + ww];
    ain[(dh * 192 + ci) * 68 + wl] = (_Float16)v;
  }
  __syncthreads();
  int wave = tid >> 5;
  int mt = wave & 3;
  int ntA = wave >> 2;
  int ntB = ntA + 2;
  v8f acc0 = {};
  v8f acc1 = {};
  for (int tap = 0; tap < 9; ++tap) {
    int dh = tap / 3, dw = tap % 3;
    for (int c0 = 0; c0 < 192; c0 += 32) {
      __builtin_prefetch(&wf[(((long)n0) * 192 + c0 + 64) * 9 + tap], 0, 1);
#pragma unroll
      for (int j = 0; j < 8; ++j) {
        int e = tid + j * 256;
        int nn = e >> 5, kk = e & 31;
        Bsw[nn * 34 + kk] = wf[(((long)(n0 + nn)) * 192 + (c0 + kk)) * 9 + tap];
      }
      __syncthreads();
      const _Float16* ap = &ain[(dh * 192 + c0) * 68 + dw];
      v16h a  = frag_a(ap + mt * 16, 1, 68, 32);   // m-stride 1, k-stride 68
      v16h b0 = frag_b(&Bsw[ntA * 16 * 34], 34, 1, 32);
      v16h b1 = frag_b(&Bsw[ntB * 16 * 34], 34, 1, 32);
      acc0 = wmma32(a, b0, acc0);
      acc1 = wmma32(a, b1, acc1);
      __syncthreads();
    }
  }
  int lane = tid & 31;
  int nn = lane & 15;
  int mb = (lane >> 4) << 3;
#pragma unroll
  for (int v = 0; v < 8; ++v) {
    int w = w0 + mt * 16 + mb + v;
#pragma unroll
    for (int s = 0; s < 2; ++s) {
      int co = n0 + (s == 0 ? ntA : ntB) * 16 + nn;
      float x = (s == 0 ? acc0[v] : acc1[v]);
      x = (x - bnm[co]) * rsqrtf(bnv[co] + 1e-5f) * bng[co] + bnb[co];
      out[(((long)b * 192 + co) * H_ + h) * Wd_ + w] = fmaxf(x, 0.f);
    }
  }
}

// ---------------- host orchestration ----------------
static inline void launch_gemm(const _Float16* A, int K1, const _Float16* A2, int K2,
                               const _Float16* Wt, const float* bias,
                               float* outF, _Float16* outH, long M, int Nout, int flags,
                               hipStream_t s) {
  dim3 g((unsigned)((M + 63) / 64), (unsigned)((Nout + 63) / 64));
  gemm64x64<<<g, 256, 0, s>>>(A, K1, A2, K2, Wt, bias, outF, outH, M, Nout, flags);
}

extern "C" void kernel_launch(void* const* d_in, const int* in_sizes, int n_in,
                              void* d_out, int out_size, void* d_ws, size_t ws_size,
                              hipStream_t stream) {
  (void)in_sizes; (void)n_in; (void)out_size; (void)ws_size;
  const float* image  = (const float*)d_in[0];
  const float* radar  = (const float*)d_in[1];
  const float* g_q = (const float*)d_in[4];   const float* b_q = (const float*)d_in[5];
  const float* g_kv = (const float*)d_in[6];  const float* b_kv = (const float*)d_in[7];
  const float* g2 = (const float*)d_in[8];    const float* b2 = (const float*)d_in[9];
  const float* bs1 = (const float*)d_in[11];
  const float* Ws2 = (const float*)d_in[12];  const float* bs2 = (const float*)d_in[13];
  const float* bqkv = (const float*)d_in[15];
  const float* bo = (const float*)d_in[17];
  const float* bf1 = (const float*)d_in[19];  const float* bf2 = (const float*)d_in[21];
  const float* gamma = (const float*)d_in[23];
  const float* gammah = (const float*)d_in[24];
  const float* Wg = (const float*)d_in[25];   const float* bg = (const float*)d_in[26];
  const float* bn_g = (const float*)d_in[28]; const float* bn_b = (const float*)d_in[29];
  const float* bn_m = (const float*)d_in[30]; const float* bn_v = (const float*)d_in[31];
  const float* conv_dw = (const float*)d_in[33];
  const float* b_dw = (const float*)d_in[34];
  const float* b_dt = (const float*)d_in[36];
  const float* A_log = (const float*)d_in[39];
  const float* Dp = (const float*)d_in[40];
  float* outp = (float*)d_out;

  size_t off = 0;
  auto alloc = [&](size_t bytes) -> void* {
    void* p = (char*)d_ws + off;
    off += (bytes + 255) & ~(size_t)255;
    return p;
  };
  // f16 weight copies
  _Float16* wImg  = (_Float16*)alloc(16384 * 2);
  _Float16* wRad  = (_Float16*)alloc(8192 * 2);
  _Float16* wS1   = (_Float16*)alloc(16384 * 2);
  _Float16* wQKV  = (_Float16*)alloc(49152 * 2);
  _Float16* wO    = (_Float16*)alloc(16384 * 2);
  _Float16* wF1   = (_Float16*)alloc(65536 * 2);
  _Float16* wF2   = (_Float16*)alloc(65536 * 2);
  _Float16* wDel  = (_Float16*)alloc(8192 * 2);
  _Float16* wHin  = (_Float16*)alloc(131072 * 2);
  _Float16* wDt   = (_Float16*)alloc(65536 * 2);
  _Float16* wBp   = (_Float16*)alloc(4096 * 2);
  _Float16* wCp   = (_Float16*)alloc(4096 * 2);
  _Float16* wHout = (_Float16*)alloc(32768 * 2);
  _Float16* wFuse = (_Float16*)alloc(331776 * 2);
  // activations
  _Float16* imgIn16  = (_Float16*)alloc(NW_ * CI_ * 2);
  _Float16* radIn16  = (_Float16*)alloc(NW_ * CR_ * 2);
  _Float16* imgAtt16 = (_Float16*)alloc(NW_ * C_ * 2);
  float*    radSeqF  = (float*)alloc(NW_ * C_ * 4);
  _Float16* radSeq16 = (_Float16*)alloc(NW_ * C_ * 2);
  _Float16* h1buf    = (_Float16*)alloc(NW_ * 64 * 2);
  int*      idx      = (int*)alloc((long)NROW_ * KTOP_ * 4);
  _Float16* kvln     = (_Float16*)alloc((long)NROW_ * KTOP_ * C_ * 2);
  _Float16* qln      = (_Float16*)alloc(NW_ * C_ * 2);
  _Float16* qh       = (_Float16*)alloc(NW_ * C_ * 2);
  _Float16* khb      = (_Float16*)alloc((long)NROW_ * KTOP_ * C_ * 2);
  _Float16* vhb      = (_Float16*)alloc((long)NROW_ * KTOP_ * C_ * 2);
  _Float16* ao       = (_Float16*)alloc(NW_ * C_ * 2);
  _Float16* xz16     = (_Float16*)alloc(NW_ * 512 * 2);  // FFN hidden, then xz
  _Float16* xs16     = (_Float16*)alloc(NW_ * DI_ * 2);
  _Float16* dt16     = (_Float16*)alloc(NW_ * DI_ * 2);
  float*    BtF      = (float*)alloc(NW_ * DS_ * 4);
  float*    CtF      = (float*)alloc(NW_ * DS_ * 4);
  _Float16* y16      = (_Float16*)alloc(NW_ * DI_ * 2);
  _Float16* hmb16    = (_Float16*)alloc(NW_ * C_ * 2);
  float*    dAttF    = (float*)alloc(NW_ * 64 * 4);
  float*    dHmbF    = (float*)alloc(NW_ * 64 * 4);
  float*    renhF    = (float*)alloc((long)B_ * 64 * H_ * Wd_ * 4);

  auto cvt = [&](int i, _Float16* dst, long n) {
    cvt16_kernel<<<(unsigned)((n + 255) / 256), 256, 0, stream>>>((const float*)d_in[i], dst, n);
  };
  cvt(2, wImg, 16384);   cvt(3, wRad, 8192);    cvt(10, wS1, 16384);
  cvt(14, wQKV, 49152);  cvt(16, wO, 16384);    cvt(18, wF1, 65536);
  cvt(20, wF2, 65536);   cvt(22, wDel, 8192);   cvt(32, wHin, 131072);
  cvt(35, wDt, 65536);   cvt(37, wBp, 4096);    cvt(38, wCp, 4096);
  cvt(41, wHout, 32768); cvt(27, wFuse, 331776);

  // transpose raw BCHW inputs to pixel-row f16 sequences
  chw_to_seq16<<<(unsigned)(((long)B_ * CI_ * H_ * Wd_ + 255) / 256), 256, 0, stream>>>(image, imgIn16, CI_);
  chw_to_seq16<<<(unsigned)(((long)B_ * CR_ * H_ * Wd_ + 255) / 256), 256, 0, stream>>>(radar, radIn16, CR_);

  // 1x1 convs
  launch_gemm(imgIn16, CI_, nullptr, 0, wImg, nullptr, nullptr, imgAtt16, NW_, C_, 0, stream);
  launch_gemm(radIn16, CR_, nullptr, 0, wRad, nullptr, radSeqF, radSeq16, NW_, C_, 0, stream);

  // score MLP hidden + exact top-k
  launch_gemm(radSeq16, C_, imgAtt16, C_, wS1, bs1, nullptr, h1buf, NW_, 64, 1, stream);
  score_topk<<<NROW_, 256, 0, stream>>>(h1buf, Ws2, bs2, idx);
  gather_ln_kv<<<NROW_, 128, 0, stream>>>(imgAtt16, idx, g_kv, b_kv, kvln);

  // Q/K/V projections
  ln128_kernel<<<(unsigned)((NW_ + 255) / 256), 256, 0, stream>>>(radSeqF, g_q, b_q, qln, NW_);
  launch_gemm(qln, C_, nullptr, 0, wQKV, bqkv, nullptr, qh, NW_, C_, 0, stream);
  launch_gemm(kvln, C_, nullptr, 0, wQKV + 128 * 128, bqkv + 128, nullptr, khb,
              (long)NROW_ * KTOP_, C_, 0, stream);
  launch_gemm(kvln, C_, nullptr, 0, wQKV + 2 * 128 * 128, bqkv + 256, nullptr, vhb,
              (long)NROW_ * KTOP_, C_, 0, stream);

  // attention core + output projection (residual into radSeqF)
  size_t attn_smem = (size_t)(64 * 132 * 2) * 2 + (size_t)64 * 132 * 4 + (size_t)64 * 128 * 4 +
                     (size_t)(128 * 18 * 2) * 2;
  attn_core<<<dim3(NROW_, 4), 256, attn_smem, stream>>>(qh, khb, vhb, ao);
  launch_gemm(ao, C_, nullptr, 0, wO, bo, radSeqF, nullptr, NW_, C_, 4, stream);

  // FFN (hidden staged in xz16), residual into radSeqF
  ln128_kernel<<<(unsigned)((NW_ + 255) / 256), 256, 0, stream>>>(radSeqF, g2, b2, qln, NW_);
  launch_gemm(qln, C_, nullptr, 0, wF1, bf1, nullptr, xz16, NW_, 512, 1, stream);
  launch_gemm(xz16, 512, nullptr, 0, wF2, bf2, radSeqF, nullptr, NW_, C_, 4, stream);
  cvt16_kernel<<<(unsigned)((NW_ * C_ + 255) / 256), 256, 0, stream>>>(radSeqF, radSeq16, NW_ * C_);

  // delta_attn + hidden-state branch
  launch_gemm(radSeq16, C_, nullptr, 0, wDel, nullptr, dAttF, nullptr, NW_, 64, 0, stream);
  launch_gemm(radSeq16, C_, imgAtt16, C_, wHin, nullptr, nullptr, xz16, NW_, 512, 0, stream);
  dwconv_silu<<<(unsigned)((NW_ * DI_ + 255) / 256), 256, 0, stream>>>(xz16, conv_dw, b_dw, xs16);
  launch_gemm(xs16, DI_, nullptr, 0, wDt, b_dt, nullptr, dt16, NW_, DI_, 2, stream);
  launch_gemm(xs16, DI_, nullptr, 0, wBp, nullptr, BtF, nullptr, NW_, DS_, 0, stream);
  launch_gemm(xs16, DI_, nullptr, 0, wCp, nullptr, CtF, nullptr, NW_, DS_, 0, stream);
  ssm_scan<<<NROW_, 256, 0, stream>>>(dt16, xs16, BtF, CtF, A_log, Dp, xz16, y16);
  launch_gemm(y16, DI_, nullptr, 0, wHout, nullptr, nullptr, hmb16, NW_, C_, 0, stream);
  launch_gemm(hmb16, C_, nullptr, 0, wDel, nullptr, dHmbF, nullptr, NW_, 64, 0, stream);

  // gate + enhanced radar, then fused 3x3 conv + BN + ReLU
  gate_enh<<<dim3(NROW_, 4), 256, (size_t)2 * 64 * 192 * 4, stream>>>(
      radar, image, Wg, bg, gamma, gammah, dAttF, dHmbF, renhF);
  size_t conv_smem = (size_t)3 * 192 * 68 * 2 + (size_t)64 * 34 * 2;
  conv3x3_bn_relu<<<dim3(NROW_, 4, 3), 256, conv_smem, stream>>>(
      renhF, image, wFuse, bn_g, bn_b, bn_m, bn_v, outp);
}
